// LSTM1_24756191494329
// MI455X (gfx1250) — matrix operand order repaired and checked by reading, not compiled, not run on hardware
//
#include <hip/hip_runtime.h>

// ---------------------------------------------------------------------------
// LSTM cell forward for MI455X (gfx1250), wave32 + v_wmma_f32_16x16x32_bf16.
//
// GEMM: all_input[B,256] @ {Wf,Wif,Wit,Wo}[256,128]  (f32 emulated via bf16
// hi/lo split, 3 WMMA products per partial -> ~f32 accuracy, f32 accumulate)
// Epilogue fused: sigmoid/tanh gates, cell-state update, both outputs.
//
// Work partition:
//   block = 256 threads = 8 waves; wave w -> state tile w (16 of 128 cols)
//   block owns 64 rows (4 M-tiles of 16 per wave) -> grid = 131072/64 = 2048
//   per wave: C accums = 4 mtiles x 4 gates x v8f = 128 VGPRs, 384 WMMAs.
//
// A tile (64x256) is cooperatively converted ONCE per block into bf16 hi/lo
// WMMA-A fragments in LDS (64 KB), then consumed via ds_load_b128 -- removes
// the 8x-redundant per-wave cvt VALU work and global A re-reads.
//
// d_ws layout: pre-swizzled weight fragments in native WMMA B layout:
//   hi: 256 frags * 512 bf16 (1 KB each) = 256 KB, then lo: 256 KB.
//   frag index = ((gate*8 + stateTile)*8 + kstep)
// ---------------------------------------------------------------------------

typedef __attribute__((ext_vector_type(16))) __bf16 v16bf;
typedef __attribute__((ext_vector_type(8)))  float  v8f;

#define NB   131072   // batch
#define SS   128      // state size
#define CATK 256      // concat K

#define WMMA_BF16(A, Bm, C) \
  __builtin_amdgcn_wmma_f32_16x16x32_bf16(false, (A), false, (Bm), (short)0, (C), false, false)

__device__ __forceinline__ float sigf(float v)       { return 1.0f / (1.0f + __expf(-v)); }
__device__ __forceinline__ float tanhf_fast(float v) { return 1.0f - 2.0f / (1.0f + __expf(2.0f * v)); }

// ---------------------------------------------------------------------------
// Prep: swizzle 4 weight matrices [256x128] f32 into bf16 hi/lo B-fragments.
// One 32-lane block per fragment (256 fragments).
// B fragment (32x16, 16-bit): lane L -> column n = L%16;
//   lanes 0-15 hold K = kstep*32 + 0..15, lanes 16-31 hold K = kstep*32 + 16..31,
//   halves h = consecutive K.
// ---------------------------------------------------------------------------
__global__ __launch_bounds__(32) void lstm_prep_weights(
    const float* __restrict__ Wf, const float* __restrict__ Wif,
    const float* __restrict__ Wit, const float* __restrict__ Wo,
    __bf16* __restrict__ hiBuf, __bf16* __restrict__ loBuf)
{
  const int frag = blockIdx.x;      // ((g*8 + t)*8 + s)
  const int lane = threadIdx.x;     // 0..31
  const int s = frag & 7;
  const int t = (frag >> 3) & 7;
  const int g = frag >> 6;
  const float* W = (g == 0) ? Wf : (g == 1) ? Wif : (g == 2) ? Wit : Wo;

  const int n  = t * 16 + (lane & 15);
  const int kb = s * 32 + ((lane >> 4) << 4);   // 0 or 16 within the k-step

  __bf16* hp = hiBuf + (size_t)frag * 512 + lane * 16;
  __bf16* lp = loBuf + (size_t)frag * 512 + lane * 16;
#pragma unroll
  for (int h = 0; h < 16; ++h) {
    float w  = W[(size_t)(kb + h) * SS + n];
    __bf16 hi = (__bf16)w;
    __bf16 lo = (__bf16)(w - (float)hi);
    hp[h] = hi;
    lp[h] = lo;
  }
}

// ---------------------------------------------------------------------------
// Main fused LSTM kernel.
// ---------------------------------------------------------------------------
__global__ __launch_bounds__(256) void lstm_main(
    const float* __restrict__ x, const float* __restrict__ prev_out,
    const float* __restrict__ prev_cell,
    const float* __restrict__ bfv_p, const float* __restrict__ bif_p,
    const float* __restrict__ bit_p, const float* __restrict__ bo_p,
    const __bf16* __restrict__ hiBuf, const __bf16* __restrict__ loBuf,
    float* __restrict__ out_state, float* __restrict__ out_y)
{
  // A fragments in LDS: slot = (mt*8 + s)*32 + lane, 16 halves per slot.
  __shared__ __align__(32) __bf16 aHi[4 * 8 * 32 * 16];   // 32 KB
  __shared__ __align__(32) __bf16 aLo[4 * 8 * 32 * 16];   // 32 KB

  const int tid   = threadIdx.x;
  const int lane  = tid & 31;
  const int t     = tid >> 5;                // wave id == state tile (0..7)
  const int rlo   = lane & 15;               // A row within tile / C column
  const int hi16  = lane >> 4;               // lane half-group
  const int rowBase = blockIdx.x * 64;       // 4 M-tiles of 16 rows

  // ---- cooperative stage: convert A tile (64 rows x 256 K) to LDS frags ----
  // 1024 lane-slots total, 4 per thread. Slot (mt,s,lane): in this code's 16-bit
// A layout, lane holds row (lane&15), one contiguous 16-element K run at
// koff + (lane>>4)*16.
#pragma unroll
  for (int j = 0; j < 4; ++j) {
    const int slot = tid + j * 256;
    const int smt  = slot >> 8;              // m-tile
    const int ssk  = (slot >> 5) & 7;        // k-step
    const int sl   = slot & 31;              // fragment lane
    const float* src  = (ssk < 4) ? x : prev_out;
    const int    koff = (ssk < 4) ? ssk * 32 : ssk * 32 - 128;
    const int    col0 = koff + (sl >> 4) * 16;
    const int    row  = rowBase + smt * 16 + (sl & 15);
    const float* p = src + (size_t)row * SS + col0;
    float4 f0 = *(const float4*)(p + 0);
    float4 f1 = *(const float4*)(p + 4);
    float4 f2 = *(const float4*)(p + 8);
    float4 f3 = *(const float4*)(p + 12);
    float r[16] = {f0.x, f0.y, f0.z, f0.w, f1.x, f1.y, f1.z, f1.w,
                   f2.x, f2.y, f2.z, f2.w, f3.x, f3.y, f3.z, f3.w};
    v16bf vh, vl;
#pragma unroll
    for (int h = 0; h < 16; ++h) {
      __bf16 hiv = (__bf16)r[h];
      vh[h] = hiv;
      vl[h] = (__bf16)(r[h] - (float)hiv);
    }
    *(v16bf*)&aHi[slot * 16] = vh;
    *(v16bf*)&aLo[slot * 16] = vl;
  }
  __syncthreads();

  v8f c[4][4];                               // [mtile][gate]
#pragma unroll
  for (int mt = 0; mt < 4; ++mt)
#pragma unroll
    for (int g = 0; g < 4; ++g)
#pragma unroll
      for (int v = 0; v < 8; ++v) c[mt][g][v] = 0.0f;

#pragma unroll 1
  for (int s = 0; s < 8; ++s) {
    v16bf ahi[4], alo[4];
#pragma unroll
    for (int mt = 0; mt < 4; ++mt) {
      const int slot = (mt * 8 + s) * 32 + lane;
      ahi[mt] = *(const v16bf*)&aHi[slot * 16];
      alo[mt] = *(const v16bf*)&aLo[slot * 16];
    }
#pragma unroll
    for (int g = 0; g < 4; ++g) {
      const size_t fi = ((size_t)(g * 8 + t) * 8 + s) * 512 + (size_t)lane * 16;
      v16bf bhi = *(const v16bf*)(hiBuf + fi);
      v16bf blo = *(const v16bf*)(loBuf + fi);
#pragma unroll
      for (int mt = 0; mt < 4; ++mt) {
        c[mt][g] = WMMA_BF16(ahi[mt], bhi, c[mt][g]);  // main term
        c[mt][g] = WMMA_BF16(alo[mt], bhi, c[mt][g]);  // A residual
        c[mt][g] = WMMA_BF16(ahi[mt], blo, c[mt][g]);  // W residual
      }
    }
  }

  // ---- fused LSTM epilogue -------------------------------------------------
  // C/D layout: VGPR v, lane L -> row M = v + (L<16?0:8), col N = L%16.
  const int n = t * 16 + rlo;
  const float bfv  = bfv_p[n];
  const float bifv = bif_p[n];
  const float bitv = bit_p[n];
  const float bov  = bo_p[n];

#pragma unroll
  for (int mt = 0; mt < 4; ++mt) {
#pragma unroll
    for (int v = 0; v < 8; ++v) {
      const int row = rowBase + mt * 16 + hi16 * 8 + v;
      const size_t idx = (size_t)row * SS + n;
      float pass_through = sigf(c[mt][0][v] + bfv);
      float input_g = sigf(c[mt][1][v] + bifv) * tanhf_fast(c[mt][2][v] + bitv);
      float output_g = sigf(c[mt][3][v] + bov);
      float cs  = pass_through * prev_cell[idx] + input_g;
      float out = tanhf_fast(cs) * output_g;
      out_state[idx] = cs;
      out_y[idx]     = out;
    }
  }
}

// ---------------------------------------------------------------------------
extern "C" void kernel_launch(void* const* d_in, const int* in_sizes, int n_in,
                              void* d_out, int out_size, void* d_ws, size_t ws_size,
                              hipStream_t stream) {
  const float* x    = (const float*)d_in[0];
  const float* po   = (const float*)d_in[1];
  const float* pc   = (const float*)d_in[2];
  const float* Wf   = (const float*)d_in[3];
  const float* bf_  = (const float*)d_in[4];
  const float* Wif  = (const float*)d_in[5];
  const float* bif_ = (const float*)d_in[6];
  const float* Wit  = (const float*)d_in[7];
  const float* bit_ = (const float*)d_in[8];
  const float* Wo   = (const float*)d_in[9];
  const float* bo_  = (const float*)d_in[10];

  __bf16* hiBuf = (__bf16*)d_ws;
  __bf16* loBuf = (__bf16*)((char*)d_ws + 256 * 1024);  // 256 KB hi, 256 KB lo

  float* out_state = (float*)d_out;
  float* out_y     = (float*)d_out + (size_t)NB * SS;

  // 256 fragments, one 32-lane block each: swizzle weights into WMMA B layout.
  lstm_prep_weights<<<256, 32, 0, stream>>>(Wf, Wif, Wit, Wo, hiBuf, loBuf);

  // 2048 blocks x 256 threads: each block = 64 rows x full 128 state cols.
  lstm_main<<<NB / 64, 256, 0, stream>>>(x, po, pc, bf_, bif_, bit_, bo_,
                                         hiBuf, loBuf, out_state, out_y);
}
